// Encoder_v0_6468220748615
// MI455X (gfx1250) — matrix operand order repaired and checked by reading, not compiled, not run on hardware
//
#include <hip/hip_runtime.h>
#include <hip/hip_bf16.h>
#include <math.h>

// ---------------------------------------------------------------------------
// MI455X / gfx1250 implementation.
//
// The reference network is fully linear (NAC layers have no activation
// between them), so it collapses to:
//   delta_uni[k] = sum_{t,c} X[k,t,c] * g[t] * v1[c]
//   y[k,c] = delta_uni[k]*wue[c] + baseline[k,c]*(1 + 1e-3*noise[k,c])
// X (134 MB) dominates traffic -> HBM roofline ~5.8us @ 23.3 TB/s.
// The X reduction is expressed as v_wmma_f32_16x16x32_bf16 accumulations
// (A = 16x32 X tile in bf16, B = g[t]*v1[:] in column 0 only).
// ---------------------------------------------------------------------------

typedef __attribute__((ext_vector_type(16))) __bf16 v16bf;
typedef __attribute__((ext_vector_type(8)))  float  v8f;

#define NKd 2048
#define NTd 512
#define NCd 32
#define NMd 512

// workspace layout (float offsets)
#define WT1E  ((size_t)0)
#define WT2E  (WT1E + 512*512)
#define WT3E  (WT2E + 512*512)
#define WD1E  (WT3E + 512*512)
#define WD2E  (WD1E + 512*512)
#define WZ1E  (WD2E + 512*512)   // 1024
#define WZ2E  (WZ1E + 1024)      // 32
#define WD3E  (WZ2E + 32)        // 512
#define WUE   (WD3E + 512)       // 32
#define V1O   (WUE  + 32)        // 32
#define AVO   (V1O  + 32)        // 512
#define DVO   (AVO  + 512)       // 512
#define H1O   (DVO  + 512)       // 512
#define H2O   (H1O  + 512)       // 512
#define GVO   (H2O  + 512)       // 512
#define DELTA (GVO  + 512)       // 2048
#define BVECO (DELTA + 2048)     // 16384 bf16 == 8192 floats

// ---- effective weights: w_eff = tanh(w) * sigmoid(m) ----------------------
__global__ void eff_kernel(float* __restrict__ dst, const float* __restrict__ w,
                           const float* __restrict__ m, int n) {
  int i = blockIdx.x * blockDim.x + threadIdx.x;
  if (i < n) {
    float s = 1.0f / (1.0f + __expf(-m[i]));
    dst[i] = tanhf(w[i]) * s;
  }
}

// ---- delta accumulator init ------------------------------------------------
__global__ void init_delta_kernel(float* __restrict__ ws) {
  int i = blockIdx.x * blockDim.x + threadIdx.x;
  if (i < NKd) ws[DELTA + i] = 0.0f;
}

// ---- collapse the linear chain into v1[32], g[512], bvec[t*32+c] ----------
__global__ __launch_bounds__(512) void chain_kernel(float* __restrict__ ws) {
  const int tid = threadIdx.x;

  // v1[c] = sum_j wz2e[j] * wz1e[j*32 + c]
  if (tid < 32) {
    float s = 0.0f;
    for (int j = 0; j < 32; ++j) s += ws[WZ2E + j] * ws[WZ1E + (size_t)j * 32 + tid];
    ws[V1O + tid] = s;
  }
  __syncthreads();

  // a[j] = sum_l wd3e[l] * wd2e[l*512 + j]
  {
    float s = 0.0f;
    for (int j = 0; j < 512; ++j) s += ws[WD3E + j] * ws[WD2E + (size_t)j * 512 + tid];
    ws[AVO + tid] = s;
  }
  __syncthreads();
  // d[m] = sum_j a[j] * wd1e[j*512 + m]
  {
    float s = 0.0f;
    for (int j = 0; j < 512; ++j) s += ws[AVO + j] * ws[WD1E + (size_t)j * 512 + tid];
    ws[DVO + tid] = s;
  }
  __syncthreads();
  // h1[m] = sum_j d[j] * wt3e[j*512 + m]
  {
    float s = 0.0f;
    for (int j = 0; j < 512; ++j) s += ws[DVO + j] * ws[WT3E + (size_t)j * 512 + tid];
    ws[H1O + tid] = s;
  }
  __syncthreads();
  // h2[m] = sum_j h1[j] * wt2e[j*512 + m]
  {
    float s = 0.0f;
    for (int j = 0; j < 512; ++j) s += ws[H1O + j] * ws[WT2E + (size_t)j * 512 + tid];
    ws[H2O + tid] = s;
  }
  __syncthreads();
  // g[t] = sum_j h2[j] * wt1e[j*512 + t]
  {
    float s = 0.0f;
    for (int j = 0; j < 512; ++j) s += ws[H2O + j] * ws[WT1E + (size_t)j * 512 + tid];
    ws[GVO + tid] = s;
  }
  __syncthreads();

  // bvec[t*32+c] = bf16( g[t] * v1[c] )
  __bf16* bv = (__bf16*)(ws + BVECO);
  for (int i = tid; i < NTd * NCd; i += 512) {
    int t = i >> 5, c = i & 31;
    bv[i] = (__bf16)(ws[GVO + t] * ws[V1O + c]);
  }
}

// ---- main streamer: delta_uni via WMMA over X ------------------------------
// grid = 128 k-tiles x 4 t-splits, block = 256 (8 waves); each wave: 16 t's.
__global__ __launch_bounds__(256) void stream_wmma_kernel(
    const float* __restrict__ X, float* __restrict__ ws) {
  const int ktile  = blockIdx.x >> 2;
  const int tsplit = blockIdx.x & 3;
  const int wave   = threadIdx.x >> 5;
  const int lane   = threadIdx.x & 31;
  const int m      = lane & 15;        // row within 16x16 tile
  const int hi     = lane >> 4;        // 0: K 0-7/16-23, 1: K 8-15/24-31

  const __bf16* __restrict__ bvec = (const __bf16*)(ws + BVECO);

  const int t0 = tsplit * 128 + wave * 16;
  const float* __restrict__ row =
      X + (size_t)(ktile * 16 + m) * (NTd * NCd) + (size_t)hi * 8;

  v8f acc0 = {}; v8f acc1 = {};

#pragma unroll 4
  for (int tt = 0; tt < 16; ++tt) {
    const int t = t0 + tt;
    const float* p = row + (size_t)t * 32;
    __builtin_prefetch(p + 512, 0, 0);   // ~2 KB ahead in the stream

    float4 a0 = *(const float4*)(p);
    float4 a1 = *(const float4*)(p + 4);
    float4 a2 = *(const float4*)(p + 16);
    float4 a3 = *(const float4*)(p + 20);

    v16bf A;
    A[0]  = (__bf16)a0.x; A[1]  = (__bf16)a0.y; A[2]  = (__bf16)a0.z; A[3]  = (__bf16)a0.w;
    A[4]  = (__bf16)a1.x; A[5]  = (__bf16)a1.y; A[6]  = (__bf16)a1.z; A[7]  = (__bf16)a1.w;
    A[8]  = (__bf16)a2.x; A[9]  = (__bf16)a2.y; A[10] = (__bf16)a2.z; A[11] = (__bf16)a2.w;
    A[12] = (__bf16)a3.x; A[13] = (__bf16)a3.y; A[14] = (__bf16)a3.z; A[15] = (__bf16)a3.w;

    // B: column N==0 holds g[t]*v1[:]; lane0 -> K0-7/16-23, lane16 -> K8-15/24-31
    const __bf16* bp = bvec + (size_t)t * 32 + (size_t)hi * 8;
    const __bf16 zero = (__bf16)0.0f;
    v16bf B;
#pragma unroll
    for (int i = 0; i < 16; ++i) {
      __bf16 bvv = bp[i < 8 ? i : i + 8];
      B[i] = (m == 0) ? bvv : zero;
    }

    if (tt & 1)
      acc1 = __builtin_amdgcn_wmma_f32_16x16x32_bf16(false, A, false, B,
                                                     (short)0, acc1, false, false);
    else
      acc0 = __builtin_amdgcn_wmma_f32_16x16x32_bf16(false, A, false, B,
                                                     (short)0, acc0, false, false);
  }

  // D column 0: lane0 holds M=0..7 in vgprs 0..7, lane16 holds M=8..15
  if (m == 0) {
    float* dst = ws + DELTA + (size_t)ktile * 16 + (size_t)hi * 8;
#pragma unroll
    for (int i = 0; i < 8; ++i) atomicAdd(dst + i, acc0[i] + acc1[i]);
  }
}

// ---- finalize: baseline + noise + delta*wue --------------------------------
// block 256 = 8 waves; each wave handles one k row (32 channels on 32 lanes).
__global__ __launch_bounds__(256) void finalize_kernel(
    const float* __restrict__ X, const float* __restrict__ noise,
    const float* __restrict__ ws, float* __restrict__ out) {
  const int wave = threadIdx.x >> 5;
  const int c    = threadIdx.x & 31;
  const int k    = blockIdx.x * 8 + wave;
  if (k >= NKd) return;

  float lx = X[(size_t)k * (NTd * NCd) + (size_t)(NTd - 1) * NCd + c];
  float s = lx;
#pragma unroll
  for (int off = 16; off >= 1; off >>= 1) s += __shfl_xor(s, off, 32);
  float baseline = lx - s * (1.0f / 32.0f);

  float d  = ws[DELTA + k];
  float wu = ws[WUE + c];
  float nz = noise[(size_t)k * 32 + c];
  out[(size_t)k * 32 + c] = d * wu + baseline + baseline * (nz * 0.001f);
}

// ---------------------------------------------------------------------------
extern "C" void kernel_launch(void* const* d_in, const int* in_sizes, int n_in,
                              void* d_out, int out_size, void* d_ws, size_t ws_size,
                              hipStream_t stream) {
  (void)in_sizes; (void)n_in; (void)out_size; (void)ws_size;

  const float* X     = (const float*)d_in[0];
  const float* noise = (const float*)d_in[1];
  float* ws  = (float*)d_ws;
  float* out = (float*)d_out;

  auto eff = [&](size_t dstOff, int wi, int mi, int n) {
    eff_kernel<<<(n + 255) / 256, 256, 0, stream>>>(
        ws + dstOff, (const float*)d_in[wi], (const float*)d_in[mi], n);
  };

  // effective weights
  eff(WT1E, 8, 9,  512 * 512);
  eff(WT2E, 10, 11, 512 * 512);
  eff(WT3E, 12, 13, 512 * 512);
  eff(WD1E, 14, 15, 512 * 512);
  eff(WD2E, 16, 17, 512 * 512);
  eff(WZ1E, 2, 3, 32 * 32);
  eff(WZ2E, 4, 5, 32);
  eff(WD3E, 18, 19, 512);
  eff(WUE,  6, 7, 32);

  init_delta_kernel<<<(NKd + 255) / 256, 256, 0, stream>>>(ws);
  chain_kernel<<<1, 512, 0, stream>>>(ws);
  stream_wmma_kernel<<<128 * 4, 256, 0, stream>>>(X, ws);
  finalize_kernel<<<NKd / 8, 256, 0, stream>>>(X, noise, ws, out);
}
